// GroupGRUCell_sharedParameters_76424648065321
// MI455X (gfx1250) — matrix operand order repaired and checked by reading, not compiled, not run on hardware
//
#include <hip/hip_runtime.h>

// Problem dims (match reference)
constexpr int kB   = 256;    // batch
constexpr int kU   = 64;     // num GRUs
constexpr int kDIN = 256;    // input size (== K for x path)
constexpr int kH   = 256;    // hidden size (== K for h path)
constexpr int kS   = 8;      // schemas
constexpr int k3H  = 3 * kH; // 768

typedef __attribute__((ext_vector_type(16))) __bf16 v16bf;
typedef __attribute__((ext_vector_type(8)))  __bf16 v8bf;
typedef __attribute__((ext_vector_type(4)))  __bf16 v4bf;
typedef __attribute__((ext_vector_type(8)))  float  v8f;

// LDS tile geometry: 16 rows x 256 bf16, row stride padded to 528 B
// (132 dwords -> bank advance 4/row; lanes 0-15 doing b128 reads cover all
// 64 banks exactly).
constexpr int kRowB   = 528;            // bytes per padded LDS row
constexpr int kTileB  = 16 * kRowB;     // 8448 B per tile

// ---------------------------------------------------------------------------
// Phase 1a: per-unit weight combine + f32 -> bf16
//   out[u][o][d] = sum_s sw[u][s] * pool[s][o][d]
// Stored N-major / K-contiguous: the WMMA B-fragment friendly layout.
// ---------------------------------------------------------------------------
__global__ __launch_bounds__(256) void combine_weights(
    const float* __restrict__ pool,  // [S, 3H, K]
    const float* __restrict__ sw,    // [U, S]
    __bf16* __restrict__ out)        // [U, 3H, K] bf16
{
    const int u   = blockIdx.y;
    const int idx = (blockIdx.x * 256 + threadIdx.x) * 4;  // within 3H*K
    const float* swu = sw + u * kS;

    float ax = 0.f, ay = 0.f, az = 0.f, aw = 0.f;
#pragma unroll
    for (int s = 0; s < kS; ++s) {
        const float4 v =
            *reinterpret_cast<const float4*>(pool + (size_t)s * k3H * kDIN + idx);
        const float w = swu[s];
        ax += w * v.x; ay += w * v.y; az += w * v.z; aw += w * v.w;
    }
    v4bf o;
    o[0] = (__bf16)ax; o[1] = (__bf16)ay; o[2] = (__bf16)az; o[3] = (__bf16)aw;
    *reinterpret_cast<v4bf*>(out + (size_t)u * k3H * kDIN + idx) = o;
}

// ---------------------------------------------------------------------------
// Phase 1b: bulk f32 -> bf16 conversion of activations (x / hidden).
// Each thread converts 8 contiguous elements.
// ---------------------------------------------------------------------------
__global__ __launch_bounds__(256) void convert_f32_bf16(
    const float* __restrict__ src, __bf16* __restrict__ dst)
{
    const int i = blockIdx.x * 256 + threadIdx.x;  // 8-element group index
    const float4* p = reinterpret_cast<const float4*>(src) + (size_t)i * 2;
    const float4 a = p[0], b = p[1];
    v8bf o;
    o[0] = (__bf16)a.x; o[1] = (__bf16)a.y; o[2] = (__bf16)a.z; o[3] = (__bf16)a.w;
    o[4] = (__bf16)b.x; o[5] = (__bf16)b.y; o[6] = (__bf16)b.z; o[7] = (__bf16)b.w;
    reinterpret_cast<v8bf*>(dst)[i] = o;
}

// ---------------------------------------------------------------------------
// Async global -> LDS copy of one 16-byte chunk (CDNA5 VGLOBAL async op).
// ldsoff: LDS byte address (low 32 bits of flat LDS pointer).
// ---------------------------------------------------------------------------
__device__ inline void async_copy_b128(unsigned ldsoff, const void* gaddr)
{
    asm volatile("global_load_async_to_lds_b128 %0, %1, off"
                 :: "v"(ldsoff), "v"(gaddr) : "memory");
}

// Fast device activations built on v_exp_f32 / v_rcp_f32 natives.
__device__ inline float fast_sigmoid(float t)
{
    return __builtin_amdgcn_rcpf(1.f + __expf(-t));
}
__device__ inline float fast_tanh(float t)
{
    const float e = __expf(-2.f * t);               // tanh = (1-e)/(1+e)
    return (1.f - e) * __builtin_amdgcn_rcpf(1.f + e);
}

// ---------------------------------------------------------------------------
// Phase 2: fused per-unit dual GEMM + GRU gates.
// grid = (B/16, H/128, U), block = 256 (8 waves).
// A-tiles (x,bf16 16x256 and hidden,bf16 16x256) staged once per block into
// LDS via async copies; all 8 waves build A-fragments from LDS.
// Each wave: one 16(batch) x 16(hidden) output tile; six 16x16 f32
// accumulators (i_r,i_i,i_n,h_r,h_i,h_n); gate math in registers.
// ---------------------------------------------------------------------------
__global__ __launch_bounds__(256) void gru_gemm(
    const __bf16* __restrict__ xb,   // [B, U, DIN] bf16
    const __bf16* __restrict__ hb,   // [B, U, H]   bf16
    const float*  __restrict__ hid,  // [B, U, H]   f32 (for epilogue)
    const __bf16* __restrict__ wx,   // [U, 3H, DIN] bf16
    const __bf16* __restrict__ wh,   // [U, 3H, H]   bf16
    float* __restrict__ out)         // [B, U, H]
{
    __shared__ __bf16 ldsX[kTileB / 2];
    __shared__ __bf16 ldsH[kTileB / 2];

    const int lane = threadIdx.x & 31;
    const int wave = threadIdx.x >> 5;
    const int lrow = lane & 15;
    const int lhi  = lane >> 4;       // 0 or 1

    const int u  = blockIdx.z;
    const int m0 = blockIdx.x * 16;                 // batch tile base
    const int h0 = blockIdx.y * 128 + wave * 16;    // hidden-col tile base

    // ---- Stage A-tiles into LDS asynchronously: 1024 x 16B chunks, 4/thread.
    const unsigned xbase = (unsigned)(size_t)&ldsX[0];
    const unsigned hbase = (unsigned)(size_t)&ldsH[0];
#pragma unroll
    for (int i = 0; i < 4; ++i) {
        const int c   = threadIdx.x + 256 * i;      // 0..1023
        const int mat = c >> 9;                     // 0 = x, 1 = hidden
        const int rc  = c & 511;
        const int row = rc >> 5;                    // 0..15
        const int col = rc & 31;                    // 16B units within row
        const __bf16* g = (mat ? hb : xb) +
            ((size_t)(m0 + row) * kU + u) * kDIN + col * 8;
        const unsigned l = (mat ? hbase : xbase) + row * kRowB + col * 16;
        async_copy_b128(l, g);
    }
    asm volatile("s_wait_asynccnt 0" ::: "memory");
    __syncthreads();

    // ---- B-fragment lane base pointers (r-gate; i/n at +256*K, +512*K).
    const __bf16* wxp = wx + ((size_t)u * k3H + (h0 + lrow)) * kDIN + lhi * 16;
    const __bf16* whp = wh + ((size_t)u * k3H + (h0 + lrow)) * kH   + lhi * 16;
    constexpr size_t gstep = (size_t)kH * kDIN;     // elements per gate group

    // A-fragment lane base within LDS: row lrow, two 16B runs at ka, ka+8.
    const __bf16* lxp = &ldsX[0] + (size_t)lrow * (kRowB / 2);
    const __bf16* lhp = &ldsH[0] + (size_t)lrow * (kRowB / 2);

    v8f axr = {}, axi = {}, axn = {};
    v8f ahr = {}, ahi = {}, ahn = {};

#pragma unroll
    for (int k0 = 0; k0 < kDIN; k0 += 32) {
        const int ka = k0 + lhi * 16;

        const v8bf xlo = *reinterpret_cast<const v8bf*>(lxp + ka);
        const v8bf xhi = *reinterpret_cast<const v8bf*>(lxp + ka + 8);
        const v8bf hlo = *reinterpret_cast<const v8bf*>(lhp + ka);
        const v8bf hhi = *reinterpret_cast<const v8bf*>(lhp + ka + 8);
        v16bf ax, ah;
#pragma unroll
        for (int t = 0; t < 8; ++t) {
            ax[t] = xlo[t]; ax[t + 8] = xhi[t];
            ah[t] = hlo[t]; ah[t + 8] = hhi[t];
        }

        const v16bf bxr = *reinterpret_cast<const v16bf*>(wxp + k0);
        const v16bf bxi = *reinterpret_cast<const v16bf*>(wxp + gstep + k0);
        const v16bf bxn = *reinterpret_cast<const v16bf*>(wxp + 2 * gstep + k0);
        const v16bf bhr = *reinterpret_cast<const v16bf*>(whp + k0);
        const v16bf bhi = *reinterpret_cast<const v16bf*>(whp + gstep + k0);
        const v16bf bhn = *reinterpret_cast<const v16bf*>(whp + 2 * gstep + k0);

        axr = __builtin_amdgcn_wmma_f32_16x16x32_bf16(false, ax, false, bxr, (short)0, axr, false, false);
        axi = __builtin_amdgcn_wmma_f32_16x16x32_bf16(false, ax, false, bxi, (short)0, axi, false, false);
        axn = __builtin_amdgcn_wmma_f32_16x16x32_bf16(false, ax, false, bxn, (short)0, axn, false, false);
        ahr = __builtin_amdgcn_wmma_f32_16x16x32_bf16(false, ah, false, bhr, (short)0, ahr, false, false);
        ahi = __builtin_amdgcn_wmma_f32_16x16x32_bf16(false, ah, false, bhi, (short)0, ahi, false, false);
        ahn = __builtin_amdgcn_wmma_f32_16x16x32_bf16(false, ah, false, bhn, (short)0, ahn, false, false);
    }

    // ---- Epilogue: GRU gate math fully in registers (native exp/rcp).
    // C/D mapping: acc[j] on this lane = (row m0 + j + 8*lhi, col h0 + lrow).
    const int hcol = h0 + lrow;
#pragma unroll
    for (int j = 0; j < 8; ++j) {
        const int b = m0 + j + lhi * 8;
        const size_t off = ((size_t)b * kU + u) * kH + hcol;
        const float hp = hid[off];
        const float r  = fast_sigmoid(axr[j] + ahr[j]);
        const float ig = fast_sigmoid(axi[j] + ahi[j]);
        const float ng = fast_tanh(axn[j] + r * ahn[j]);
        out[off] = ng + ig * (hp - ng);
    }
}

// ---------------------------------------------------------------------------
extern "C" void kernel_launch(void* const* d_in, const int* in_sizes, int n_in,
                              void* d_out, int out_size, void* d_ws, size_t ws_size,
                              hipStream_t stream)
{
    (void)in_sizes; (void)n_in; (void)out_size; (void)ws_size;
    const float* x      = (const float*)d_in[0];
    const float* hidden = (const float*)d_in[1];
    const float* pool_x = (const float*)d_in[2];
    const float* pool_h = (const float*)d_in[3];
    const float* sw_x   = (const float*)d_in[4];
    const float* sw_h   = (const float*)d_in[5];
    float* out = (float*)d_out;

    // Scratch layout (bf16): wxb 25.2MB | whb 25.2MB | xb 8.4MB | hb 8.4MB
    __bf16* wxb = (__bf16*)d_ws;
    __bf16* whb = wxb + (size_t)kU * k3H * kDIN;
    __bf16* xb  = whb + (size_t)kU * k3H * kH;
    __bf16* hb  = xb  + (size_t)kB * kU * kDIN;

    dim3 g1((k3H * kDIN) / (256 * 4), kU, 1);          // 192 x 64
    combine_weights<<<g1, 256, 0, stream>>>(pool_x, sw_x, wxb);
    combine_weights<<<g1, 256, 0, stream>>>(pool_h, sw_h, whb);

    const int nconv = (kB * kU * kDIN) / (256 * 8);    // 2048 blocks
    convert_f32_bf16<<<nconv, 256, 0, stream>>>(x, xb);
    convert_f32_bf16<<<nconv, 256, 0, stream>>>(hidden, hb);

    dim3 g2(kB / 16, kH / 128, kU);                    // 16 x 2 x 64
    gru_gemm<<<g2, 256, 0, stream>>>(xb, hb, hidden, wxb, whb, out);
}
